// AutoencoderDecoderLayer_24970939859152
// MI455X (gfx1250) — matrix operand order repaired and checked by reading, not compiled, not run on hardware
//
#include <hip/hip_runtime.h>

// ---------------------------------------------------------------------------
// Transformer decoder layer (post-LN, causal) for MI455X / gfx1250.
// - All large GEMMs: v_wmma_f32_16x16x32_bf16 (bf16 in, f32 accumulate)
// - Global->LDS staging: CDNA5 async copies (global_load_async_to_lds_b128,
//   ASYNCcnt) + LDS double buffering
// - Attention: flash-style online softmax, never materializes SxS scores
// ---------------------------------------------------------------------------

#define S_    1024
#define B_    8
#define E_    1024
#define NH_   16
#define HD_   64
#define F_    4096
#define MTOK  (S_*B_)   // 8192 token rows

typedef __attribute__((ext_vector_type(16))) __bf16 bf16x16;
typedef __attribute__((ext_vector_type(8)))  float  v8f;

union FragU { bf16x16 v; uint4 q[2]; };

// ---- CDNA5 async global->LDS copy (16B per lane, tracked by ASYNCcnt) ----
__device__ __forceinline__ void async_copy_b128(const void* gsrc, void* ldst) {
  const unsigned int lofs = (unsigned int)(size_t)ldst;  // low 32b of generic LDS addr = LDS byte offset
  asm volatile("global_load_async_to_lds_b128 %0, %1, off"
               :: "v"(lofs), "v"(gsrc) : "memory");
}
__device__ __forceinline__ void wait_async0() {
#if __has_builtin(__builtin_amdgcn_s_wait_asynccnt)
  __builtin_amdgcn_s_wait_asynccnt(0);
#else
  asm volatile("s_wait_asynccnt 0x0" ::: "memory");
#endif
}

// A-matrix fragment (16x32, M x K), CDNA5 VGPR layout:
// lane r=lane&15 owns row M=r; kh=lane>>4 selects K-half.
// VGPRs 0-3: K = kh*8 + 0..7 (16B contiguous), VGPRs 4-7: K = 16 + kh*8 + 0..7.
__device__ __forceinline__ bf16x16 load_a_frag(const __bf16* smem, int row0, int ld,
                                               int kofs, int lane) {
  const int r  = row0 + (lane & 15);
  const int kh = (lane >> 4) * 8;
  const __bf16* p = smem + r * ld + kofs + kh;
  FragU f;
  f.q[0] = *(const uint4*)(p);
  f.q[1] = *(const uint4*)(p + 16);
  return f.v;
}

// B-matrix fragment (32x16, K x N) where memory holds B^T row-major
// (row n, col k): lane n=lane&15 owns column N=n; VGPRs 0-3: K = kh*8 + 0..7, VGPRs 4-7: K = 16 + kh*8 + 0..7.
__device__ __forceinline__ bf16x16 load_b_frag(const __bf16* smem, int col0, int ld,
                                               int kofs, int lane) {
  const int n  = col0 + (lane & 15);
  const int kh = (lane >> 4) * 8;
  const __bf16* p = smem + n * ld + kofs + kh;
  FragU f;
  f.q[0] = *(const uint4*)(p);
  f.q[1] = *(const uint4*)(p + 16);
  return f.v;
}

__device__ __forceinline__ v8f wmma_bf16(bf16x16 a, bf16x16 b, v8f c) {
  return __builtin_amdgcn_wmma_f32_16x16x32_bf16(false, a, false, b, (short)0, c,
                                                 false, false);
}

// ---------------------------------------------------------------------------
// GEMM: C[m][n] = act( (sum_k A[m][k]*W[n][k]) + bias[n] ) * scale
// A: MxK bf16 row-major.  W: NxK bf16 row-major (torch Linear weight).
// 128x128 tile / workgroup, BK=64, 8 waves; wave (mw = w&3, nw = w>>2)
// computes a 32x64 strip: 8 accumulators, 8 WMMAs per 32-deep k-step.
// LDS double-buffered; next chunk streams in via async copies while the
// current chunk is consumed by WMMA.
// ---------------------------------------------------------------------------
#define GT   128  // M/N tile
#define GK   64   // K chunk
#define GLD  72   // padded LDS row stride (144B: 16B aligned, bank-friendly)

__global__ __launch_bounds__(256)
void gemm_bf16_wmma(const __bf16* __restrict__ A, const __bf16* __restrict__ W,
                    const float* __restrict__ bias,
                    float* __restrict__ Cf, __bf16* __restrict__ Cb,
                    int M, int N, int K, float scale, int relu) {
  __shared__ __bf16 As[2][GT * GLD];
  __shared__ __bf16 Bs[2][GT * GLD];
  const int tid  = threadIdx.x;
  const int wv   = tid >> 5;
  const int lane = tid & 31;
  const int mw   = (wv & 3) * 32;       // wave row base within tile
  const int nw   = (wv >> 2) * 64;      // wave col base within tile
  const int tm   = blockIdx.y * GT;
  const int tn   = blockIdx.x * GT;

  v8f acc[2][4] = {};

  // issue async copies for one 128x64 A tile + 128x64 B tile (8 per thread)
  auto load_tile = [&](int buf, int kc) {
#pragma unroll
    for (int i = 0; i < 4; ++i) {
      const int idx = tid + i * 256;        // 1024 uint4 per matrix
      const int r = idx >> 3, c4 = (idx & 7) * 8;
      async_copy_b128(A + (size_t)(tm + r) * K + kc + c4, &As[buf][r * GLD + c4]);
      async_copy_b128(W + (size_t)(tn + r) * K + kc + c4, &Bs[buf][r * GLD + c4]);
    }
  };

  load_tile(0, 0);
  wait_async0();
  __syncthreads();

  int buf = 0;
  for (int kc = 0; kc < K; kc += GK) {
    const int nxt = kc + GK;
    if (nxt < K) load_tile(buf ^ 1, nxt);     // stream next chunk (no wait yet)

    const __bf16* Asb = &As[buf][0];
    const __bf16* Bsb = &Bs[buf][0];
#pragma unroll
    for (int ks = 0; ks < 2; ++ks) {
      bf16x16 af0 = load_a_frag(Asb, mw,      GLD, ks * 32, lane);
      bf16x16 af1 = load_a_frag(Asb, mw + 16, GLD, ks * 32, lane);
#pragma unroll
      for (int nt = 0; nt < 4; ++nt) {
        bf16x16 bfr = load_b_frag(Bsb, nw + nt * 16, GLD, ks * 32, lane);
        acc[0][nt] = wmma_bf16(af0, bfr, acc[0][nt]);
        acc[1][nt] = wmma_bf16(af1, bfr, acc[1][nt]);
      }
    }
    if (nxt < K) {
      wait_async0();       // our async stores into buf^1 have landed
      __syncthreads();     // everyone finished reading buf & writing buf^1
      buf ^= 1;
    }
  }

  // Epilogue. C/D layout: element (M = j + 8*(lane>>4), N = lane&15).
  const int hi = lane >> 4, nl = lane & 15;
#pragma unroll
  for (int ms = 0; ms < 2; ++ms) {
#pragma unroll
    for (int nt = 0; nt < 4; ++nt) {
      const int n = tn + nw + nt * 16 + nl;
      const float bvv = bias ? bias[n] : 0.0f;
#pragma unroll
      for (int j = 0; j < 8; ++j) {
        const int m = tm + mw + ms * 16 + j + 8 * hi;
        float v = (acc[ms][nt][j] + bvv) * scale;
        if (relu) v = fmaxf(v, 0.0f);
        if (Cf) Cf[(size_t)m * N + n] = v;
        if (Cb) Cb[(size_t)m * N + n] = (__bf16)v;
      }
    }
  }
}

// ---------------------------------------------------------------------------
// Flash attention (causal). Q pre-scaled by 1/sqrt(HD). Layout (S,B,E):
// element (s,b,h,d) at ptr[((s*B)+b)*E + h*HD + d].
// Grid: (S/64, NH, B); 128 threads = 4 waves; wave w owns q-rows [16w,16w+16).
// ---------------------------------------------------------------------------
#define FLD 72

__global__ __launch_bounds__(128)
void flash_attn_wmma(const __bf16* __restrict__ Q, const __bf16* __restrict__ Kg,
                     const __bf16* __restrict__ Vg, __bf16* __restrict__ O) {
  __shared__ __bf16 Qs [64 * FLD];
  __shared__ __bf16 Ks [64 * FLD];
  __shared__ __bf16 Vts[64 * FLD];   // transposed: row = d, col = key
  __shared__ __bf16 Ps [64 * FLD];
  __shared__ float  Ss [64 * 65];
  __shared__ float  mrow[64], lrow[64], arow[64];

  const int qt = blockIdx.x, h = blockIdx.y, b = blockIdx.z;
  const int tid = threadIdx.x, wv = tid >> 5, lane = tid & 31;
  const size_t hofs = (size_t)h * HD_;

#pragma unroll
  for (int i = 0; i < 4; ++i) {             // Q tile via async copies
    const int idx = tid + i * 128;
    const int r = idx >> 3, c4 = (idx & 7) * 8;
    const size_t g = ((size_t)(qt * 64 + r) * B_ + b) * E_ + hofs + c4;
    async_copy_b128(Q + g, &Qs[r * FLD + c4]);
  }
  if (tid < 64) { mrow[tid] = -3.0e38f; lrow[tid] = 0.0f; }

  v8f oacc[4] = {};

  for (int j = 0; j <= qt; ++j) {
    __syncthreads();   // previous iteration done reading Ks/Vts/Ps
    // K tile: async copies; V tile: batched loads + in-register transpose
    uint4 vreg[4];
#pragma unroll
    for (int i = 0; i < 4; ++i) {
      const int idx = tid + i * 128;
      const int r = idx >> 3, c4 = (idx & 7) * 8;
      const size_t g = ((size_t)(j * 64 + r) * B_ + b) * E_ + hofs + c4;
      async_copy_b128(Kg + g, &Ks[r * FLD + c4]);
      vreg[i] = *(const uint4*)(Vg + g);
    }
#pragma unroll
    for (int i = 0; i < 4; ++i) {
      const int idx = tid + i * 128;
      const int r = idx >> 3, c4 = (idx & 7) * 8;
      const __bf16* ve = (const __bf16*)&vreg[i];
#pragma unroll
      for (int e = 0; e < 8; ++e) Vts[(c4 + e) * FLD + r] = ve[e];
    }
    wait_async0();
    __syncthreads();

    // S = Q @ K^T  (M=64 q-rows, N=64 keys, K=HD=64)
    v8f sacc[4] = {};
#pragma unroll
    for (int ks = 0; ks < 2; ++ks) {
      bf16x16 af = load_a_frag(Qs, wv * 16, FLD, ks * 32, lane);
#pragma unroll
      for (int nt = 0; nt < 4; ++nt)
        sacc[nt] = wmma_bf16(af, load_b_frag(Ks, nt * 16, FLD, ks * 32, lane), sacc[nt]);
    }
    {
      const int hi = lane >> 4, nl = lane & 15;
#pragma unroll
      for (int nt = 0; nt < 4; ++nt)
#pragma unroll
        for (int jj = 0; jj < 8; ++jj) {
          const int m  = wv * 16 + jj + 8 * hi;
          const int nn = nt * 16 + nl;
          const float v = sacc[nt][jj];
          Ss[m * 65 + nn] = ((j * 64 + nn) <= (qt * 64 + m)) ? v : -1.0e9f;
        }
    }
    __syncthreads();

    // Online softmax: one thread per q-row.
    if (tid < 64) {
      const int r = tid;
      const float mo = mrow[r];
      float mx = mo;
      for (int nn = 0; nn < 64; ++nn) mx = fmaxf(mx, Ss[r * 65 + nn]);
      const float alpha = __expf(mo - mx);
      float ps = 0.0f;
      for (int nn = 0; nn < 64; ++nn) {
        const float p = __expf(Ss[r * 65 + nn] - mx);
        Ps[r * FLD + nn] = (__bf16)p;
        ps += p;
      }
      lrow[r] = lrow[r] * alpha + ps;
      mrow[r] = mx;
      arow[r] = alpha;
    }
    __syncthreads();

    // O = O*alpha + P @ V
    {
      const int hi = lane >> 4;
#pragma unroll
      for (int nt = 0; nt < 4; ++nt)
#pragma unroll
        for (int jj = 0; jj < 8; ++jj)
          oacc[nt][jj] *= arow[wv * 16 + jj + 8 * hi];
    }
#pragma unroll
    for (int ks = 0; ks < 2; ++ks) {
      bf16x16 af = load_a_frag(Ps, wv * 16, FLD, ks * 32, lane);
#pragma unroll
      for (int nt = 0; nt < 4; ++nt)
        oacc[nt] = wmma_bf16(af, load_b_frag(Vts, nt * 16, FLD, ks * 32, lane), oacc[nt]);
    }
  }
  __syncthreads();
  {
    const int hi = lane >> 4, nl = lane & 15;
#pragma unroll
    for (int nt = 0; nt < 4; ++nt)
#pragma unroll
      for (int jj = 0; jj < 8; ++jj) {
        const int m = wv * 16 + jj + 8 * hi;
        const float inv = 1.0f / lrow[m];
        const size_t g = ((size_t)(qt * 64 + m) * B_ + b) * E_ + hofs + nt * 16 + nl;
        O[g] = (__bf16)(oacc[nt][jj] * inv);
      }
  }
}

// ---------------------------------------------------------------------------
// Block-wide double reduction (sum, sumsq) — wave32 shuffles.
// ---------------------------------------------------------------------------
__device__ __forceinline__ void block_reduce2(float& a, float& b, float* redA, float* redB) {
#pragma unroll
  for (int off = 16; off > 0; off >>= 1) {
    a += __shfl_down(a, off, 32);
    b += __shfl_down(b, off, 32);
  }
  const int wv = threadIdx.x >> 5, lane = threadIdx.x & 31;
  if (lane == 0) { redA[wv] = a; redB[wv] = b; }
  __syncthreads();
  if (wv == 0) {
    float ta = (lane < 8) ? redA[lane] : 0.0f;
    float tb = (lane < 8) ? redB[lane] : 0.0f;
#pragma unroll
    for (int off = 4; off > 0; off >>= 1) {
      ta += __shfl_down(ta, off, 32);
      tb += __shfl_down(tb, off, 32);
    }
    if (lane == 0) { redA[0] = ta; redB[0] = tb; }
  }
  __syncthreads();
  a = redA[0]; b = redB[0];
}

// out = LN(res + y); optional bf16 copy.  One block per token row, E=1024.
__global__ __launch_bounds__(256)
void ln_residual(const float* __restrict__ res, const float* __restrict__ y,
                 const float* __restrict__ g, const float* __restrict__ beta,
                 float* __restrict__ outf, __bf16* __restrict__ outb) {
  __shared__ float redA[8], redB[8];
  const size_t row = blockIdx.x;
  float vals[4], s = 0.0f, sq = 0.0f;
#pragma unroll
  for (int i = 0; i < 4; ++i) {
    const int e = threadIdx.x + i * 256;
    const float v = res[row * E_ + e] + y[row * E_ + e];
    vals[i] = v; s += v; sq += v * v;
  }
  block_reduce2(s, sq, redA, redB);
  const float mu   = s * (1.0f / E_);
  const float var  = sq * (1.0f / E_) - mu * mu;
  const float rstd = rsqrtf(var + 1e-5f);
#pragma unroll
  for (int i = 0; i < 4; ++i) {
    const int e = threadIdx.x + i * 256;
    const float o = (vals[i] - mu) * rstd * g[e] + beta[e];
    outf[row * E_ + e] = o;
    if (outb) outb[row * E_ + e] = (__bf16)o;
  }
}

// x2 = LN(x1 + sigmoid(g1 + zg[b]) * zv[b]); writes f32 + bf16.
__global__ __launch_bounds__(256)
void gate_ln(const float* __restrict__ x1, const float* __restrict__ g1,
             const float* __restrict__ zg, const float* __restrict__ zv,
             const float* __restrict__ g, const float* __restrict__ beta,
             float* __restrict__ outf, __bf16* __restrict__ outb) {
  __shared__ float redA[8], redB[8];
  const size_t row = blockIdx.x;
  const int b = (int)(row % B_);            // row = s*B + b
  float vals[4], s = 0.0f, sq = 0.0f;
#pragma unroll
  for (int i = 0; i < 4; ++i) {
    const int e = threadIdx.x + i * 256;
    const float t = g1[row * E_ + e] + zg[(size_t)b * E_ + e];
    const float gate = 1.0f / (1.0f + __expf(-t));
    const float v = x1[row * E_ + e] + gate * zv[(size_t)b * E_ + e];
    vals[i] = v; s += v; sq += v * v;
  }
  block_reduce2(s, sq, redA, redB);
  const float mu   = s * (1.0f / E_);
  const float var  = sq * (1.0f / E_) - mu * mu;
  const float rstd = rsqrtf(var + 1e-5f);
#pragma unroll
  for (int i = 0; i < 4; ++i) {
    const int e = threadIdx.x + i * 256;
    const float o = (vals[i] - mu) * rstd * g[e] + beta[e];
    outf[row * E_ + e] = o;
    outb[row * E_ + e] = (__bf16)o;
  }
}

// Tiny z projections: out[b][n] = z[b]·W[n] + bias[n].  (B=8 rows, f32.)
__global__ __launch_bounds__(256)
void zproj(const float* __restrict__ z, const float* __restrict__ W,
           const float* __restrict__ bias, float* __restrict__ out) {
  const int n = blockIdx.x * 256 + threadIdx.x;
  const int b = blockIdx.y;
  const float4* zr = (const float4*)(z + (size_t)b * E_);
  const float4* wr = (const float4*)(W + (size_t)n * E_);
  float s = bias[n];
  for (int k = 0; k < E_ / 4; ++k) {
    const float4 a = zr[k], w = wr[k];
    s += a.x * w.x + a.y * w.y + a.z * w.z + a.w * w.w;
  }
  out[(size_t)b * E_ + n] = s;
}

__global__ void cvt_f32_bf16(const float* __restrict__ in, __bf16* __restrict__ out,
                             size_t n) {
  size_t i = (size_t)blockIdx.x * blockDim.x + threadIdx.x;
  const size_t stride = (size_t)gridDim.x * blockDim.x;
  for (; i < n; i += stride) out[i] = (__bf16)in[i];
}

// ---------------------------------------------------------------------------
extern "C" void kernel_launch(void* const* d_in, const int* in_sizes, int n_in,
                              void* d_out, int out_size, void* d_ws, size_t ws_size,
                              hipStream_t stream) {
  (void)in_sizes; (void)n_in; (void)out_size; (void)ws_size;

  const float* x    = (const float*)d_in[0];
  const float* z    = (const float*)d_in[1];
  const float* wq   = (const float*)d_in[2];
  const float* bq   = (const float*)d_in[3];
  const float* wk   = (const float*)d_in[4];
  const float* bk   = (const float*)d_in[5];
  const float* wv   = (const float*)d_in[6];
  const float* bv   = (const float*)d_in[7];
  const float* wo   = (const float*)d_in[8];
  const float* bo   = (const float*)d_in[9];
  const float* ln1g = (const float*)d_in[10];
  const float* ln1b = (const float*)d_in[11];
  const float* pghw = (const float*)d_in[12];
  const float* pghb = (const float*)d_in[13];
  const float* pgzw = (const float*)d_in[14];
  const float* pgzb = (const float*)d_in[15];
  const float* pvw  = (const float*)d_in[16];
  const float* pvb  = (const float*)d_in[17];
  const float* ln2g = (const float*)d_in[18];
  const float* ln2b = (const float*)d_in[19];
  const float* fc1w = (const float*)d_in[20];
  const float* fc1b = (const float*)d_in[21];
  const float* fc2w = (const float*)d_in[22];
  const float* fc2b = (const float*)d_in[23];
  const float* ln3g = (const float*)d_in[24];
  const float* ln3b = (const float*)d_in[25];
  float* out = (float*)d_out;

  // --- workspace carve-out ---
  char* ws = (char*)d_ws;
  size_t cur = 0;
  auto alloc = [&](size_t bytes) -> char* {
    char* p = ws + cur;
    cur = (cur + bytes + 255) & ~(size_t)255;
    return p;
  };
  const size_t EE = (size_t)E_ * E_;
  const size_t ME = (size_t)MTOK * E_;
  __bf16* wq_b   = (__bf16*)alloc(EE * 2);
  __bf16* wk_b   = (__bf16*)alloc(EE * 2);
  __bf16* wv_b   = (__bf16*)alloc(EE * 2);
  __bf16* wo_b   = (__bf16*)alloc(EE * 2);
  __bf16* pgh_wb = (__bf16*)alloc(EE * 2);
  __bf16* fc1_wb = (__bf16*)alloc((size_t)F_ * E_ * 2);
  __bf16* fc2_wb = (__bf16*)alloc((size_t)E_ * F_ * 2);
  __bf16* x_b    = (__bf16*)alloc(ME * 2);
  __bf16* q_b    = (__bf16*)alloc(ME * 2);
  __bf16* k_b    = (__bf16*)alloc(ME * 2);
  __bf16* v_b    = (__bf16*)alloc(ME * 2);
  __bf16* attn_b = (__bf16*)alloc(ME * 2);
  float*  tmpf   = (float*)alloc(ME * 4);
  __bf16* h_b    = (__bf16*)alloc((size_t)MTOK * F_ * 2);
  float*  x1f    = (float*)alloc(ME * 4);
  float*  x2f    = (float*)alloc(ME * 4);
  float*  zgbuf  = (float*)alloc((size_t)B_ * E_ * 4);
  float*  zvbuf  = (float*)alloc((size_t)B_ * E_ * 4);
  __bf16* x1b    = x_b;   // x_b dead after QKV projections
  __bf16* x2b    = q_b;   // q_b dead after attention

  // --- f32 -> bf16 conversions ---
  cvt_f32_bf16<<<2048, 256, 0, stream>>>(x,    x_b,    ME);
  cvt_f32_bf16<<<512,  256, 0, stream>>>(wq,   wq_b,   EE);
  cvt_f32_bf16<<<512,  256, 0, stream>>>(wk,   wk_b,   EE);
  cvt_f32_bf16<<<512,  256, 0, stream>>>(wv,   wv_b,   EE);
  cvt_f32_bf16<<<512,  256, 0, stream>>>(wo,   wo_b,   EE);
  cvt_f32_bf16<<<512,  256, 0, stream>>>(pghw, pgh_wb, EE);
  cvt_f32_bf16<<<2048, 256, 0, stream>>>(fc1w, fc1_wb, (size_t)F_ * E_);
  cvt_f32_bf16<<<2048, 256, 0, stream>>>(fc2w, fc2_wb, (size_t)E_ * F_);

  const dim3 blk(256);
  const dim3 gEE(E_ / GT, MTOK / GT);   // (8,64)
  const dim3 gEF(F_ / GT, MTOK / GT);   // (32,64)

  // --- self attention ---
  gemm_bf16_wmma<<<gEE, blk, 0, stream>>>(x_b, wq_b, bq, nullptr, q_b,
                                          MTOK, E_, E_, 0.125f /*1/sqrt(HD)*/, 0);
  gemm_bf16_wmma<<<gEE, blk, 0, stream>>>(x_b, wk_b, bk, nullptr, k_b,
                                          MTOK, E_, E_, 1.0f, 0);
  gemm_bf16_wmma<<<gEE, blk, 0, stream>>>(x_b, wv_b, bv, nullptr, v_b,
                                          MTOK, E_, E_, 1.0f, 0);
  flash_attn_wmma<<<dim3(S_ / 64, NH_, B_), dim3(128), 0, stream>>>(q_b, k_b, v_b, attn_b);
  gemm_bf16_wmma<<<gEE, blk, 0, stream>>>(attn_b, wo_b, bo, tmpf, nullptr,
                                          MTOK, E_, E_, 1.0f, 0);
  ln_residual<<<MTOK, 256, 0, stream>>>(x, tmpf, ln1g, ln1b, x1f, x1b);

  // --- gated fusion ---
  zproj<<<dim3(E_ / 256, B_), 256, 0, stream>>>(z, pgzw, pgzb, zgbuf);
  zproj<<<dim3(E_ / 256, B_), 256, 0, stream>>>(z, pvw,  pvb,  zvbuf);
  gemm_bf16_wmma<<<gEE, blk, 0, stream>>>(x1b, pgh_wb, pghb, tmpf, nullptr,
                                          MTOK, E_, E_, 1.0f, 0);
  gate_ln<<<MTOK, 256, 0, stream>>>(x1f, tmpf, zgbuf, zvbuf, ln2g, ln2b, x2f, x2b);

  // --- FFN ---
  gemm_bf16_wmma<<<gEF, blk, 0, stream>>>(x2b, fc1_wb, fc1b, nullptr, h_b,
                                          MTOK, F_, E_, 1.0f, 1 /*relu*/);
  gemm_bf16_wmma<<<gEE, blk, 0, stream>>>(h_b, fc2_wb, fc2b, tmpf, nullptr,
                                          MTOK, E_, F_, 1.0f, 0);
  ln_residual<<<MTOK, 256, 0, stream>>>(x2f, tmpf, ln3g, ln3b, out, nullptr);
}
